// MesaNetBlock_55903294325379
// MI455X (gfx1250) — matrix operand order repaired and checked by reading, not compiled, not run on hardware
//
#include <hip/hip_runtime.h>

typedef unsigned short u16;
typedef unsigned int   u32;

#define B_   8
#define T_   4096
#define D_   512
#define HID_ 1024
#define RHID_ 64

// ---------------- bf16 helpers ----------------
__device__ __forceinline__ u16 f2bf(float x) {
    u32 u = __builtin_bit_cast(u32, x);
    u += 0x7FFFu + ((u >> 16) & 1u);           // round-to-nearest-even
    return (u16)(u >> 16);
}
__device__ __forceinline__ float bf2f(u16 h) {
    u32 u = ((u32)h) << 16;
    return __builtin_bit_cast(float, u);
}

// ---------------- WMMA types ----------------
typedef __bf16 v16bf __attribute__((ext_vector_type(16)));
typedef float  v8f   __attribute__((ext_vector_type(8)));
typedef float  v2f   __attribute__((ext_vector_type(2)));

union ABu { v16bf v; uint4 q[2]; };

// A-operand 16x32 bf16 tile, row-major [M][K] with row stride `stride` (u16 units).
// ISA 7.12.2: lanes 0-15 M=lane, lanes 16-31 M=lane-16; VGPR v holds
// K = 16*(v>>2) + 2*(v&3) + (laneHi?8:0) and K+1  ->  two contiguous b128 loads.
__device__ __forceinline__ v16bf loadA(const u16* base, int stride, int lane) {
    int m  = lane & 15;
    int kh = (lane >> 4) << 3;                  // 0 or 8
    const u16* p = base + m * stride + kh;
    ABu r;
    r.q[0] = *(const uint4*)(p);                // K = kh .. kh+7
    r.q[1] = *(const uint4*)(p + 16);           // K = kh+16 .. kh+23
    return r.v;
}

// B-operand 32x16 bf16 tile stored TRANSPOSED row-major [N][K] with row stride.
// lanes 0-15 N=lane, lanes 16-31 N=lane-16; same K order as loadA:
// K = kh .. kh+7 and kh+16 .. kh+23 (kh = 0 or 8)  ->  two b128 loads.
__device__ __forceinline__ v16bf loadBT(const u16* base, int stride, int lane) {
    int n  = lane & 15;
    int kh = (lane >> 4) << 3;                  // 0 or 8
    const u16* p = base + n * stride + kh;
    ABu r;
    r.q[0] = *(const uint4*)(p);
    r.q[1] = *(const uint4*)(p + 16);
    return r.v;
}

__device__ __forceinline__ v8f bwmma(v16bf a, v16bf b, v8f c) {
    return __builtin_amdgcn_wmma_f32_16x16x32_bf16(false, a, false, b, (short)0, c, false, false);
}

__device__ __forceinline__ v8f fwmma(v2f a, v2f b, v8f c) {
#if __has_builtin(__builtin_amdgcn_wmma_f32_16x16x4_f32)
    return __builtin_amdgcn_wmma_f32_16x16x4_f32(false, a, false, b, (short)0, c, false, false);
#else
    c[0] += a.x * b.x + a.y * b.y;              // compile fallback only
    return c;
#endif
}

// ---------------- transpose+convert f32 -> bf16 : out[b][C][R] = in[b][R][C] ----
__global__ __launch_bounds__(256) void k_trans(const float* __restrict__ in,
                                               u16* __restrict__ out, int R, int C) {
    __shared__ float tile[32][33];
    long ib = (long)blockIdx.z * R * C;
    int c0 = blockIdx.x * 32, r0 = blockIdx.y * 32;
    for (int i = threadIdx.x; i < 32 * 32; i += 256) {
        int rr = i >> 5, cc = i & 31;
        tile[rr][cc] = in[ib + (long)(r0 + rr) * C + c0 + cc];
    }
    __syncthreads();
    for (int i = threadIdx.x; i < 32 * 32; i += 256) {
        int cc = i >> 5, rr = i & 31;
        out[ib + (long)(c0 + cc) * R + r0 + rr] = f2bf(tile[rr][cc]);
    }
}

// ---------------- bf16 transpose: out[b][C][R] = in[b][R][C] ------------------
__global__ __launch_bounds__(256) void k_transb(const u16* __restrict__ in,
                                                u16* __restrict__ out, int R, int C) {
    __shared__ u16 tile[32][33];
    long ib = (long)blockIdx.z * R * C;
    int c0 = blockIdx.x * 32, r0 = blockIdx.y * 32;
    for (int i = threadIdx.x; i < 32 * 32; i += 256) {
        int rr = i >> 5, cc = i & 31;
        tile[rr][cc] = in[ib + (long)(r0 + rr) * C + c0 + cc];
    }
    __syncthreads();
    for (int i = threadIdx.x; i < 32 * 32; i += 256) {
        int cc = i >> 5, rr = i & 31;
        out[ib + (long)(c0 + cc) * R + r0 + rr] = tile[rr][cc];
    }
}

// ---------------- fused feature net: 16-row tile -----------------------------
// x(16x512) -> h=gelu(x@W1+b1)(16x1024, LDS) -> f=h@W2+b2 -> LayerNorm -> f_b
__global__ __launch_bounds__(256) void k_feature(
    const float* __restrict__ x, const float* __restrict__ b1,
    const float* __restrict__ b2, const float* __restrict__ ln_g,
    const float* __restrict__ ln_b, const u16* __restrict__ w1t,
    const u16* __restrict__ w2t, u16* __restrict__ f_b) {
    constexpr int SH = 1032;                    // h-tile row stride (u16)
    constexpr int SX = 520;                     // x/f-tile row stride (u16)
    __shared__ u16 sH[16 * SH];                 // 33,024 B
    __shared__ u16 sXF[16 * SX];                // 16,640 B (x, then f)
    int tid = threadIdx.x, lane = tid & 31, w = tid >> 5;
    long row0 = (long)blockIdx.x * 16;          // global row in [0, B*T)

    // stage x tile -> bf16 LDS
    const float* xr = x + row0 * D_;
    for (int i = tid; i < 16 * D_; i += 256) {
        int r = i >> 9, c = i & 511;
        sXF[r * SX + c] = f2bf(xr[r * D_ + c]);
    }
    __syncthreads();

    // h = gelu(x @ W1 + b1): each wave does 8 of the 64 n-tiles
    for (int nt = 0; nt < 8; ++nt) {
        int ng = (w * 8 + nt) * 16;
        v8f acc = {};
        for (int kb = 0; kb < 16; ++kb) {
            v16bf a  = loadA(sXF + kb * 32, SX, lane);
            v16bf bb = loadBT(w1t + (long)ng * D_ + kb * 32, D_, lane);
            acc = bwmma(a, bb, acc);
        }
        int n = lane & 15, mh = (lane >> 4) << 3;
        float bias = b1[ng + n];
        #pragma unroll
        for (int v = 0; v < 8; ++v) {
            float val = acc[v] + bias;
            val = 0.5f * val * (1.0f + erff(val * 0.70710678118f)); // exact GELU
            sH[(v + mh) * SH + ng + n] = f2bf(val);
        }
    }
    __syncthreads();

    // f = h @ W2 + b2: each wave does 4 of the 32 n-tiles, result into sXF
    for (int nt = 0; nt < 4; ++nt) {
        int ng = (w * 4 + nt) * 16;
        v8f acc = {};
        for (int kb = 0; kb < 32; ++kb) {
            v16bf a  = loadA(sH + kb * 32, SH, lane);
            v16bf bb = loadBT(w2t + (long)ng * HID_ + kb * 32, HID_, lane);
            acc = bwmma(a, bb, acc);
        }
        int n = lane & 15, mh = (lane >> 4) << 3;
        float bias = b2[ng + n];
        #pragma unroll
        for (int v = 0; v < 8; ++v)
            sXF[(v + mh) * SX + ng + n] = f2bf(acc[v] + bias);
    }
    __syncthreads();

    // LayerNorm: 16 threads per row
    int r = tid >> 4, cl = tid & 15;
    float s = 0.f, sq = 0.f;
    for (int j = 0; j < 32; ++j) {
        float v = bf2f(sXF[r * SX + cl + j * 16]);
        s += v; sq += v * v;
    }
    for (int off = 8; off; off >>= 1) {
        s += __shfl_xor(s, off, 16);
        sq += __shfl_xor(sq, off, 16);
    }
    float mu = s * (1.0f / D_);
    float var = sq * (1.0f / D_) - mu * mu;
    float rstd = rsqrtf(var + 1e-5f);
    for (int j = 0; j < 32; ++j) {
        int c = cl + j * 16;
        float v = bf2f(sXF[r * SX + c]);
        v = (v - mu) * rstd * ln_g[c] + ln_b[c];
        f_b[(row0 + r) * D_ + c] = f2bf(v);
    }
}

// ---------------- mean over T of f -------------------------------------------
__global__ __launch_bounds__(256) void k_colmean(const u16* __restrict__ f_b,
                                                 float* __restrict__ mean) {
    int b = blockIdx.x >> 1;
    int d = ((blockIdx.x & 1) << 8) + threadIdx.x;
    const u16* fb = f_b + (long)b * T_ * D_;
    float s = 0.f;
    for (int t = 0; t < T_; ++t) s += bf2f(fb[(long)t * D_ + d]);
    mean[b * D_ + d] = s * (1.0f / T_);
}

// ---------------- reg net (tiny) ---------------------------------------------
__global__ void k_reg(const float* __restrict__ mean, const float* __restrict__ Wr1,
                      const float* __restrict__ br1, const float* __restrict__ Wr2,
                      const float* __restrict__ br2, float* __restrict__ reg) {
    __shared__ float red[RHID_];
    int b = blockIdx.x, j = threadIdx.x;        // 64 threads
    const float* mb = mean + b * D_;
    float s = br1[j];
    for (int d = 0; d < D_; ++d) s += mb[d] * Wr1[d * RHID_ + j];
    red[j] = fmaxf(s, 0.f) * Wr2[j];
    __syncthreads();
    if (j == 0) {
        float tot = br2[0];
        for (int i = 0; i < RHID_; ++i) tot += red[i];
        reg[b] = (tot > 20.f) ? tot : log1pf(expf(tot));  // softplus
    }
}

// ---------------- C[b][m][n] = sum_k P[b][m][k]*Q[b][n][k] (bf16 WMMA) --------
// 128x64 block tile, K staged 64 deep; each wave owns a 16x64 strip (A reuse
// across 4 N-tiles -> 8 WMMA per barrier pair). Optional reg[b] on diagonal.
__global__ __launch_bounds__(256) void k_gemm_nt(
    const u16* __restrict__ P, long pBatch, const u16* __restrict__ Q, long qBatch,
    float* __restrict__ C, long cBatch, int K, int ldC, const float* __restrict__ reg) {
    constexpr int ST = 72;
    __shared__ u16 sP[128 * ST];                // 18,432 B
    __shared__ u16 sQ[64 * ST];                 //  9,216 B
    int tid = threadIdx.x, lane = tid & 31, w = tid >> 5;
    int b = blockIdx.z;
    const u16* Pb = P + (long)b * pBatch + (long)(blockIdx.x * 128) * K;
    const u16* Qb = Q + (long)b * qBatch + (long)(blockIdx.y * 64) * K;
    int prow = tid >> 1, pcol = (tid & 1) * 32; // 128 rows x 64 cols, 32 u16/thread
    int qrow = tid >> 2, qcol = (tid & 3) * 16; //  64 rows x 64 cols, 16 u16/thread
    v8f acc[4] = {};
    for (int kc = 0; kc < K; kc += 64) {
        __syncthreads();
        {   // stage P (4x uint4) and Q (2x uint4), prefetch next chunk
            const uint4* sp = (const uint4*)(Pb + (long)prow * K + kc + pcol);
            uint4*       dp = (uint4*)(sP + prow * ST + pcol);
            dp[0] = sp[0]; dp[1] = sp[1]; dp[2] = sp[2]; dp[3] = sp[3];
            const uint4* sq = (const uint4*)(Qb + (long)qrow * K + kc + qcol);
            uint4*       dq = (uint4*)(sQ + qrow * ST + qcol);
            dq[0] = sq[0]; dq[1] = sq[1];
            if (kc + 64 < K) {                  // global_prefetch_b8
                __builtin_prefetch(Pb + (long)prow * K + kc + 64 + pcol, 0, 1);
                __builtin_prefetch(Qb + (long)qrow * K + kc + 64 + qcol, 0, 1);
            }
        }
        __syncthreads();
        #pragma unroll
        for (int h = 0; h < 2; ++h) {           // two 32-deep sub-chunks
            v16bf a = loadA(sP + (w * 16) * ST + h * 32, ST, lane);
            #pragma unroll
            for (int j = 0; j < 4; ++j) {
                v16bf bb = loadBT(sQ + (j * 16) * ST + h * 32, ST, lane);
                acc[j] = bwmma(a, bb, acc[j]);
            }
        }
    }
    int n = lane & 15, mh = (lane >> 4) << 3;
    int gm0 = blockIdx.x * 128 + w * 16;
    float* Cb = C + (long)b * cBatch;
    float rv = reg ? reg[b] : 0.f;
    #pragma unroll
    for (int j = 0; j < 4; ++j) {
        #pragma unroll
        for (int v = 0; v < 8; ++v) {
            int m  = gm0 + v + mh;
            int nn = blockIdx.y * 64 + j * 16 + n;
            float o = acc[j][v];
            if (reg && m == nn) o += rv;
            Cb[(long)m * ldC + nn] = o;
        }
    }
}

// ---------------- CG init: x=0, r=p=rhs --------------------------------------
__global__ __launch_bounds__(256) void k_cg_init(const float* __restrict__ rhs,
                                                 float* __restrict__ xk,
                                                 float* __restrict__ r,
                                                 float* __restrict__ p) {
    long i = (long)blockIdx.x * 256 + threadIdx.x;
    float v = rhs[i];
    xk[i] = 0.f; r[i] = v; p[i] = v;
}

// ---------------- Ap[b][e][i] = sum_j p[b][e][j] * A[b][i][j]  (f32 WMMA) -----
__global__ __launch_bounds__(256) void k_cg_matmul(const float* __restrict__ p,
                                                   const float* __restrict__ A,
                                                   float* __restrict__ Ap) {
    constexpr int ST = 66;
    __shared__ float sP[64 * ST], sA[64 * ST];  // 33,792 B total
    int tid = threadIdx.x, lane = tid & 31, w = tid >> 5;
    int b = blockIdx.z;
    const float* Pb = p + ((long)b * D_ + blockIdx.x * 64) * D_;
    const float* Ab = A + ((long)b * D_ + blockIdx.y * 64) * D_;
    int sr = w >> 1, sc0 = (w & 1) * 2;
    v8f acc0 = {}, acc1 = {};
    for (int kc = 0; kc < D_; kc += 64) {
        __syncthreads();
        for (int i = tid; i < 64 * 64; i += 256) {
            int row = i >> 6, c = i & 63;
            sP[row * ST + c] = Pb[(long)row * D_ + kc + c];
            sA[row * ST + c] = Ab[(long)row * D_ + kc + c];
        }
        __syncthreads();
        int m  = sr * 16 + (lane & 15);
        int n0 = sc0 * 16 + (lane & 15);
        int k0 = (lane >> 4) << 1;              // f32 A 16x4: laneHi -> K+2
        #pragma unroll
        for (int kk = 0; kk < 64; kk += 4) {
            v2f a;  a.x  = sP[m * ST + kk + k0];         a.y  = sP[m * ST + kk + k0 + 1];
            v2f b0; b0.x = sA[n0 * ST + kk + k0];        b0.y = sA[n0 * ST + kk + k0 + 1];
            v2f b1; b1.x = sA[(n0 + 16) * ST + kk + k0]; b1.y = sA[(n0 + 16) * ST + kk + k0 + 1];
            acc0 = fwmma(a, b0, acc0);
            acc1 = fwmma(a, b1, acc1);
        }
    }
    int n = lane & 15, mh = (lane >> 4) << 3;
    float* Ob = Ap + ((long)b * D_ + blockIdx.x * 64) * D_;
    #pragma unroll
    for (int v = 0; v < 8; ++v) {
        int m = sr * 16 + v + mh;
        Ob[(long)m * D_ + blockIdx.y * 64 + sc0 * 16 + n]       = acc0[v];
        Ob[(long)m * D_ + blockIdx.y * 64 + (sc0 + 1) * 16 + n] = acc1[v];
    }
}

// ---------------- fused CG vector update (one block per (b,e) row) -----------
__global__ __launch_bounds__(256) void k_cg_update(float* __restrict__ xk,
                                                   float* __restrict__ r,
                                                   float* __restrict__ p,
                                                   const float* __restrict__ Ap) {
    __shared__ float sbuf[16];
    long base = (long)blockIdx.x * D_;
    int tid = threadIdx.x, w = tid >> 5;
    float rv[2], pv[2], av[2], rr = 0.f, pap = 0.f;
    #pragma unroll
    for (int i = 0; i < 2; ++i) {
        int c = tid + i * 256;
        rv[i] = r[base + c]; pv[i] = p[base + c]; av[i] = Ap[base + c];
        rr += rv[i] * rv[i]; pap += pv[i] * av[i];
    }
    for (int off = 16; off; off >>= 1) {
        rr += __shfl_xor(rr, off, 32);
        pap += __shfl_xor(pap, off, 32);
    }
    if ((tid & 31) == 0) { sbuf[w] = rr; sbuf[8 + w] = pap; }
    __syncthreads();
    rr = 0.f; pap = 0.f;
    for (int i = 0; i < 8; ++i) { rr += sbuf[i]; pap += sbuf[8 + i]; }
    float alpha = rr / (pap + 1e-8f);
    float xn[2], rn[2], rn2 = 0.f;
    #pragma unroll
    for (int i = 0; i < 2; ++i) {
        xn[i] = xk[base + tid + i * 256] + alpha * pv[i];
        rn[i] = rv[i] - alpha * av[i];
        rn2 += rn[i] * rn[i];
    }
    for (int off = 16; off; off >>= 1) rn2 += __shfl_xor(rn2, off, 32);
    __syncthreads();
    if ((tid & 31) == 0) sbuf[w] = rn2;
    __syncthreads();
    rn2 = 0.f;
    for (int i = 0; i < 8; ++i) rn2 += sbuf[i];
    float beta = rn2 / (rr + 1e-8f);
    #pragma unroll
    for (int i = 0; i < 2; ++i) {
        int c = tid + i * 256;
        xk[base + c] = xn[i];
        r[base + c] = rn[i];
        p[base + c] = rn[i] + beta * pv[i];
    }
}

// ---------------- f32 -> bf16 convert ----------------------------------------
__global__ __launch_bounds__(256) void k_f2bf(const float* __restrict__ in,
                                              u16* __restrict__ out) {
    long i = (long)blockIdx.x * 256 + threadIdx.x;
    out[i] = f2bf(in[i]);
}

// ---------------- host side --------------------------------------------------
extern "C" void kernel_launch(void* const* d_in, const int* in_sizes, int n_in,
                              void* d_out, int out_size, void* d_ws, size_t ws_size,
                              hipStream_t stream) {
    const float* x      = (const float*)d_in[0];
    const float* target = (const float*)d_in[1];
    const float* W1     = (const float*)d_in[2];
    const float* b1     = (const float*)d_in[3];
    const float* W2     = (const float*)d_in[4];
    const float* b2     = (const float*)d_in[5];
    const float* ln_g   = (const float*)d_in[6];
    const float* ln_b   = (const float*)d_in[7];
    const float* Wr1    = (const float*)d_in[8];
    const float* br1    = (const float*)d_in[9];
    const float* Wr2    = (const float*)d_in[10];
    const float* br2    = (const float*)d_in[11];
    float* out = (float*)d_out;

    unsigned char* ws = (unsigned char*)d_ws;
    size_t off = 0;
    auto take = [&](size_t bytes) -> void* {
        void* pp = ws + off;
        off = (off + bytes + 255) & ~(size_t)255;
        return pp;
    };
    const long BTD = (long)B_ * T_ * D_;        // 16,777,216
    const long BDD = (long)B_ * D_ * D_;        //  2,097,152
    u16*   f_b  = (u16*)take(BTD * 2);
    u16*   fT   = (u16*)take(BTD * 2);
    u16*   tgtT = (u16*)take(BTD * 2);
    u16*   w1t  = (u16*)take((long)HID_ * D_ * 2);
    u16*   w2t  = (u16*)take((long)D_ * HID_ * 2);
    float* mean = (float*)take((long)B_ * D_ * 4);
    float* regv = (float*)take(256);
    float* Agr  = (float*)take(BDD * 4);
    float* rhs  = (float*)take(BDD * 4);
    float* xk   = (float*)take(BDD * 4);
    float* rr   = (float*)take(BDD * 4);
    float* pp   = (float*)take(BDD * 4);
    float* Apv  = (float*)take(BDD * 4);
    u16*   xk_b = (u16*)take(BDD * 2);

    // weight / target transposition + bf16 conversion
    k_trans<<<dim3(HID_ / 32, D_ / 32, 1), 256, 0, stream>>>(W1, w1t, D_, HID_);
    k_trans<<<dim3(D_ / 32, HID_ / 32, 1), 256, 0, stream>>>(W2, w2t, HID_, D_);
    k_trans<<<dim3(D_ / 32, T_ / 32, B_), 256, 0, stream>>>(target, tgtT, T_, D_);

    // feature net + LayerNorm -> f (bf16), then coalesced transpose -> fT
    k_feature<<<(B_ * T_) / 16, 256, 0, stream>>>(x, b1, b2, ln_g, ln_b, w1t, w2t, f_b);
    k_transb<<<dim3(D_ / 32, T_ / 32, B_), 256, 0, stream>>>(f_b, fT, T_, D_);

    // reg scalar per batch
    k_colmean<<<2 * B_, 256, 0, stream>>>(f_b, mean);
    k_reg<<<B_, RHID_, 0, stream>>>(mean, Wr1, br1, Wr2, br2, regv);

    // Gram (+reg*I) and rhs
    k_gemm_nt<<<dim3(4, 8, B_), 256, 0, stream>>>(fT, (long)D_ * T_, fT, (long)D_ * T_,
                                                  Agr, (long)D_ * D_, T_, D_, regv);
    k_gemm_nt<<<dim3(4, 8, B_), 256, 0, stream>>>(tgtT, (long)D_ * T_, fT, (long)D_ * T_,
                                                  rhs, (long)D_ * D_, T_, D_, (const float*)nullptr);

    // CG: 4 fixed steps in f32
    k_cg_init<<<(int)(BDD / 256), 256, 0, stream>>>(rhs, xk, rr, pp);
    for (int s = 0; s < 4; ++s) {
        k_cg_matmul<<<dim3(8, 8, B_), 256, 0, stream>>>(pp, Agr, Apv);
        k_cg_update<<<B_ * D_, 256, 0, stream>>>(xk, rr, pp, Apv);
    }

    // out[b][t][e] = sum_d f[t,d] * xk[e,d]
    k_f2bf<<<(int)(BDD / 256), 256, 0, stream>>>(xk, xk_b);
    k_gemm_nt<<<dim3(T_ / 128, 8, B_), 256, 0, stream>>>(f_b, (long)T_ * D_, xk_b, (long)D_ * D_,
                                                         out, (long)T_ * D_, D_, D_,
                                                         (const float*)nullptr);
    (void)in_sizes; (void)n_in; (void)out_size; (void)ws_size;
}
